// Edgewise_30176440222423
// MI455X (gfx1250) — matrix operand order repaired and checked by reading, not compiled, not run on hardware
//
#include <hip/hip_runtime.h>

typedef unsigned short u16;
typedef unsigned int   u32;
typedef __attribute__((ext_vector_type(16))) __bf16 v16bf;
typedef __attribute__((ext_vector_type(2)))  __bf16 v2bf;
typedef __attribute__((ext_vector_type(8)))  float  v8f;
typedef __attribute__((ext_vector_type(4)))  float  v4f;
typedef __attribute__((ext_vector_type(8)))  u32    v8u;
typedef __attribute__((ext_vector_type(4)))  u32    v4u;

// ---------------- scalar helpers ----------------
// f32 -> bf16 through the hardware converter (v_cvt_pk_bf16_f32).
__device__ __forceinline__ u16 f2bfu(float f){
  return __builtin_bit_cast(u16, (__bf16)f);
}
__device__ __forceinline__ float bfu2f(u16 h){
  return __builtin_bit_cast(float, (u32)h << 16);
}
__device__ __forceinline__ u32 pk2(float a, float b){
  v2bf p; p[0] = (__bf16)a; p[1] = (__bf16)b;
  return __builtin_bit_cast(u32, p);
}
__device__ __forceinline__ float sigmoidf_(float x){ return 1.f / (1.f + __expf(-x)); }
__device__ __forceinline__ v8f zero8(){
  v8f z;
#pragma unroll
  for(int i=0;i<8;++i) z[i]=0.f;
  return z;
}
__device__ __forceinline__ v8f wmma_bf16(v16bf a, v16bf b, v8f c){
  return __builtin_amdgcn_wmma_f32_16x16x32_bf16(false, a, false, b, (short)0, c, false, false);
}

// ---------------- fragment loaders ----------------
// A fragment (16x32 bf16, M=edges): lane&15 = row, VGPR v -> K pair per CDNA5 A layout.
// Memory layout [edge][features], K contiguous -> two 16B loads per lane.
__device__ __forceinline__ v16bf a_frag(const u16* x, long stride, int ebase, int off, int kt, int lane){
  const u16* p = x + (long)(ebase + (lane & 15)) * stride + off + kt*32 + ((lane >> 4) * 8);
  v4u lo = *(const v4u*)p;
  v4u hi = *(const v4u*)(p + 16);
  v8u u; u[0]=lo[0];u[1]=lo[1];u[2]=lo[2];u[3]=lo[3];u[4]=hi[0];u[5]=hi[1];u[6]=hi[2];u[7]=hi[3];
  return __builtin_bit_cast(v16bf, u);
}
// A fragment converted on the fly from fp32 source.
__device__ __forceinline__ v16bf a_frag_f32(const float* x, long stride, int ebase, int off, int kt, int lane){
  const float* p = x + (long)(ebase + (lane & 15)) * stride + off + kt*32 + ((lane >> 4) * 8);
  v4f a0 = *(const v4f*)p,      a1 = *(const v4f*)(p + 4);
  v4f b0 = *(const v4f*)(p+16), b1 = *(const v4f*)(p + 20);
  v8u u;
  u[0]=pk2(a0[0],a0[1]); u[1]=pk2(a0[2],a0[3]); u[2]=pk2(a1[0],a1[1]); u[3]=pk2(a1[2],a1[3]);
  u[4]=pk2(b0[0],b0[1]); u[5]=pk2(b0[2],b0[3]); u[6]=pk2(b1[0],b1[1]); u[7]=pk2(b1[2],b1[3]);
  return __builtin_bit_cast(v16bf, u);
}
// B fragment from prepacked weights: 512 bf16 per 16x32 tile, lane's 16 values contiguous.
__device__ __forceinline__ v16bf b_frag(const u16* pack, int chunk, int lane){
  const u16* p = pack + (long)chunk*512 + lane*16;
  v4u lo = *(const v4u*)p;
  v4u hi = *(const v4u*)(p + 8);
  v8u u; u[0]=lo[0];u[1]=lo[1];u[2]=lo[2];u[3]=lo[3];u[4]=hi[0];u[5]=hi[1];u[6]=hi[2];u[7]=hi[3];
  return __builtin_bit_cast(v16bf, u);
}
// B fragment from an LDS tile stored transposed as [col][32 K] bf16.
__device__ __forceinline__ v16bf b_frag_ldsT(const u16* t, int ntile, int lane){
  const u16* p = t + (long)(ntile*16 + (lane & 15)) * 32 + ((lane >> 4) * 8);
  v4u lo = *(const v4u*)p;
  v4u hi = *(const v4u*)(p + 16);
  v8u u; u[0]=lo[0];u[1]=lo[1];u[2]=lo[2];u[3]=lo[3];u[4]=hi[0];u[5]=hi[1];u[6]=hi[2];u[7]=hi[3];
  return __builtin_bit_cast(v16bf, u);
}

// D tile (16x16 f32) -> per-wave LDS transpose -> 16B bf16 global stores.
__device__ __forceinline__ void store_tile_bf16(float* stage, const v8f& acc, u16* dst,
                                                long ldDst, int lane, int validRows){
  int col = lane & 15, rofs = (lane >> 4) * 8;
#pragma unroll
  for(int r=0;r<8;++r) stage[(rofs + r)*16 + col] = acc[r];
  asm volatile("s_wait_dscnt 0" ::: "memory");
  int row = lane >> 1, cb = (lane & 1) * 8;
  if(row < validRows){
    const float* src = stage + row*16 + cb;
    v4u o;
#pragma unroll
    for(int i=0;i<4;++i) o[i] = pk2(src[2*i], src[2*i+1]);
    *(v4u*)(dst + (long)row*ldDst + cb) = o;
  }
  asm volatile("s_wait_dscnt 0" ::: "memory");
}

// Same, but multiplies by the per-(row,channel) radial weights while the data is
// still f32.  rowbase+row is the harmonic row; rad row map: 0-4->0-4, 5-12->5-8,
// 13-18->9-11.  radbase already includes e*1536 + ntile*16.
__device__ __forceinline__ void store_tile_bf16_rad(float* stage, const v8f& acc, u16* dst,
                                                    long ldDst, int lane, int validRows,
                                                    const u16* radbase, int rowbase){
  int col = lane & 15, rofs = (lane >> 4) * 8;
#pragma unroll
  for(int r=0;r<8;++r) stage[(rofs + r)*16 + col] = acc[r];
  asm volatile("s_wait_dscnt 0" ::: "memory");
  int row = lane >> 1, cb = (lane & 1) * 8;
  if(row < validRows){
    int grow   = rowbase + row;
    int radrow = (grow < 9) ? grow : ((grow < 16) ? grow - 4 : grow - 7);
    v4u rv = *(const v4u*)(radbase + (long)radrow*128 + cb);
    const float* src = stage + row*16 + cb;
    v4u o;
#pragma unroll
    for(int i=0;i<4;++i){
      float r0 = bfu2f((u16)(rv[i] & 0xffffu));
      float r1 = bfu2f((u16)(rv[i] >> 16));
      o[i] = pk2(src[2*i]*r0, src[2*i+1]*r1);
    }
    *(v4u*)(dst + (long)row*ldDst + cb) = o;
  }
  asm volatile("s_wait_dscnt 0" ::: "memory");
}

// GEMM accumulate over NB output tiles: acc[j] += Aslice @ Bpack[ntBase+j]
template<int NB>
__device__ __forceinline__ void accum_gemm(v8f* acc, const u16* x, long xs,
                                           int ebase, int xoff,
                                           const u16* pack, int KT, int ntBase, int lane){
  for(int kt=0; kt<KT; ++kt){
    v16bf a = a_frag(x, xs, ebase, xoff, kt, lane);
#pragma unroll
    for(int j=0;j<NB;++j)
      acc[j] = wmma_bf16(a, b_frag(pack, (ntBase + j)*KT + kt, lane), acc[j]);
  }
}

// ---------------- kernels ----------------
__global__ void zero_f32(float* p, long n){
  long i = (long)blockIdx.x * blockDim.x + threadIdx.x;
  if(i < n) p[i] = 0.f;
}

// fp32 [K][N] (N contiguous) -> packed bf16 B fragments (optionally negated)
__global__ void prepack_b(const float* __restrict__ w, u16* __restrict__ dst,
                          int K, int N, float sign){
  int t = blockIdx.x * 256 + threadIdx.x;
  int total = (K * N) >> 1;
  if(t >= total) return;
  int chunk = t >> 8, r = t & 255, lane = r >> 3, v = r & 7;
  int KT = K >> 5;
  int kt = chunk % KT, nt = chunk / KT;
  int k0  = kt*32 + ((v & 4) ? 16 : 0) + 2*(v & 3) + ((lane & 16) ? 8 : 0);
  int col = nt*16 + (lane & 15);
  float f0 = sign * w[(long)k0      * N + col];
  float f1 = sign * w[(long)(k0+1)  * N + col];
  ((u32*)dst)[t] = pk2(f0, f1);
}

// rad = silu(ee @ rw1 + rb1) @ rw2 + rb2  -> bf16 [E][1536]
__global__ __launch_bounds__(64) void rad_kernel(const float* __restrict__ ee,
    const float* __restrict__ b1, const float* __restrict__ b2,
    const u16* __restrict__ pw1, const u16* __restrict__ pw2,
    u16* __restrict__ rad, int E){
  __shared__ alignas(16) u16  hid[2][16*64];
  __shared__ alignas(16) float stage[2][256];
  int wv = threadIdx.x >> 5, lane = threadIdx.x & 31;
  int ebase = (blockIdx.x*2 + wv) * 16;
  if(ebase >= E) return;
  int col = lane & 15, rofs = (lane >> 4) * 8;

  v8f acc[4];
#pragma unroll
  for(int j=0;j<4;++j) acc[j] = zero8();
  for(int kt=0; kt<4; ++kt){
    v16bf a = a_frag_f32(ee, 128, ebase, 0, kt, lane);
#pragma unroll
    for(int j=0;j<4;++j) acc[j] = wmma_bf16(a, b_frag(pw1, j*4 + kt, lane), acc[j]);
  }
#pragma unroll
  for(int j=0;j<4;++j){
    float bias = b1[j*16 + col];
#pragma unroll
    for(int r=0;r<8;++r){
      float y = acc[j][r] + bias;
      y = y * sigmoidf_(y);
      hid[wv][(rofs + r)*64 + j*16 + col] = f2bfu(y);
    }
  }
  asm volatile("s_wait_dscnt 0" ::: "memory");

  v16bf a0 = a_frag(hid[wv], 64, 0, 0, 0, lane);
  v16bf a1 = a_frag(hid[wv], 64, 0, 0, 1, lane);
  for(int ng=0; ng<24; ++ng){
#pragma unroll
    for(int j=0;j<4;++j){
      int nt = ng*4 + j;
      v8f a2 = zero8();
      a2 = wmma_bf16(a0, b_frag(pw2, nt*2 + 0, lane), a2);
      a2 = wmma_bf16(a1, b_frag(pw2, nt*2 + 1, lane), a2);
      float bias = b2[nt*16 + col];
#pragma unroll
      for(int r=0;r<8;++r) a2[r] += bias;
      store_tile_bf16(stage[wv], a2, rad + (long)ebase*1536 + nt*16, 1536, lane, 16);
    }
  }
}

// per-edge: msgA = rad_scale( wigner (19x25) @ concat(nf[s],nf[r]) (25x128) ), bf16 out
__global__ __launch_bounds__(128) void gather_wigner_kernel(
    const float* __restrict__ nf, const float* __restrict__ wig,
    const int* __restrict__ snd, const int* __restrict__ rcv,
    const u16* __restrict__ rad,
    u16* __restrict__ msgA, int E){
  __shared__ alignas(16) u16  XT[4][128*32];
  __shared__ alignas(16) float stage[4][256];
  int wv = threadIdx.x >> 5, lane = threadIdx.x & 31;
  int e = blockIdx.x*4 + wv;
  if(e >= E) return;
  int s = snd[e], rr = rcv[e];
  for(int idx = lane; idx < 128*32; idx += 32){
    int c = idx >> 5, k = idx & 31;
    float v = 0.f;
    if(k < 25){
      int node = (c < 64) ? s : rr;
      v = nf[(long)node*1600 + k*64 + (c & 63)];
    }
    XT[wv][idx] = f2bfu(v);
  }
  asm volatile("s_wait_dscnt 0" ::: "memory");

  v16bf a[2];
  for(int mt=0; mt<2; ++mt){
    int row = mt*16 + (lane & 15);
    int kb  = (lane >> 4) * 8;
    const float* wrow = wig + (long)e*475 + row*25;
    v8u u;
#pragma unroll
    for(int vv=0; vv<8; ++vv){
      int k0 = ((vv < 4) ? 0 : 16) + kb + 2*(vv & 3);
      float f0 = (row < 19 && k0     < 25) ? wrow[k0]     : 0.f;
      float f1 = (row < 19 && k0 + 1 < 25) ? wrow[k0 + 1] : 0.f;
      u[vv] = pk2(f0, f1);
    }
    a[mt] = __builtin_bit_cast(v16bf, u);
  }
  const u16* radE = rad + (long)e*1536;
  for(int nt=0; nt<8; ++nt){
    v16bf b = b_frag_ldsT(XT[wv], nt, lane);
    for(int mt=0; mt<2; ++mt){
      v8f acc = zero8();
      acc = wmma_bf16(a[mt], b, acc);
      store_tile_bf16_rad(stage[wv], acc,
                          msgA + (long)e*2432 + (long)(mt*16)*128 + nt*16,
                          128, lane, (mt == 0) ? 16 : 3,
                          radE + nt*16, mt*16);
    }
  }
}

// SO2 conv #1 + gate activation: msgA[E][2432] (pre-scaled by rad) -> msgB[E][1216]
__global__ __launch_bounds__(64) void conv1_kernel(
    const u16* __restrict__ msgA,
    const u16* pk_m0, const u16* pk_r1, const u16* pk_i1, const u16* pk_i1n,
    const u16* pk_r2, const u16* pk_i2, const u16* pk_i2n,
    const float* __restrict__ b1, u16* __restrict__ msgB, int E){
  __shared__ alignas(16) u16  gate[2][16*256];
  __shared__ alignas(16) float stage[2][256];
  int wv = threadIdx.x >> 5, lane = threadIdx.x & 31;
  int ebase = (blockIdx.x*2 + wv) * 16;
  if(ebase >= E) return;
  int col = lane & 15, rofs = (lane >> 4) * 8;
  const long F1 = 2432, F2 = 1216;

  // gating = sigmoid(y0 cols 320..575)
  for(int ng=0; ng<4; ++ng){
    v8f acc[4];
#pragma unroll
    for(int j=0;j<4;++j) acc[j] = zero8();
    accum_gemm<4>(acc, msgA, F1, ebase, 0, pk_m0, 20, 20 + ng*4, lane);
#pragma unroll
    for(int j=0;j<4;++j){
      int nt = 20 + ng*4 + j;
      float bias = b1[nt*16 + col];
      int gc = (nt - 20)*16 + col;
#pragma unroll
      for(int r=0;r<8;++r)
        gate[wv][(rofs + r)*256 + gc] = f2bfu(sigmoidf_(acc[j][r] + bias));
    }
  }
  asm volatile("s_wait_dscnt 0" ::: "memory");

  // m=0 rows 0..4 (row0 silu, rows 1..4 gated)
  for(int ng=0; ng<5; ++ng){
    v8f acc[4];
#pragma unroll
    for(int j=0;j<4;++j) acc[j] = zero8();
    accum_gemm<4>(acc, msgA, F1, ebase, 0, pk_m0, 20, ng*4, lane);
#pragma unroll
    for(int j=0;j<4;++j){
      int nt = ng*4 + j;
      float bias = b1[nt*16 + col];
      int row = nt >> 2, hb = (nt & 3)*16;
      if(row == 0){
#pragma unroll
        for(int r=0;r<8;++r){ float y = acc[j][r] + bias; acc[j][r] = y * sigmoidf_(y); }
      } else {
        int gc = (row - 1)*64 + hb + col;
#pragma unroll
        for(int r=0;r<8;++r)
          acc[j][r] = (acc[j][r] + bias) * bfu2f(gate[wv][(rofs + r)*256 + gc]);
      }
      store_tile_bf16(stage[wv], acc[j], msgB + (long)ebase*F2 + row*64 + hb, F2, lane, 16);
    }
  }
  // m=1: yr rows 5..8, yi rows 9..12
  for(int half=0; half<2; ++half){
    for(int ng=0; ng<4; ++ng){
      v8f acc[4];
#pragma unroll
      for(int j=0;j<4;++j) acc[j] = zero8();
      if(half == 0){ // yr = xr@wr - xi@wi
        accum_gemm<4>(acc, msgA, F1, ebase,  640, pk_r1,  16, ng*4, lane);
        accum_gemm<4>(acc, msgA, F1, ebase, 1152, pk_i1n, 16, ng*4, lane);
      } else {       // yi = xi@wr + xr@wi
        accum_gemm<4>(acc, msgA, F1, ebase, 1152, pk_r1,  16, ng*4, lane);
        accum_gemm<4>(acc, msgA, F1, ebase,  640, pk_i1,  16, ng*4, lane);
      }
#pragma unroll
      for(int j=0;j<4;++j){
        int nt = ng*4 + j;
        int row = (half ? 9 : 5) + (nt >> 2), hb = (nt & 3)*16;
        int gc = ((half ? row - 9 : row - 5))*64 + hb + col;
#pragma unroll
        for(int r=0;r<8;++r)
          acc[j][r] *= bfu2f(gate[wv][(rofs + r)*256 + gc]);
        store_tile_bf16(stage[wv], acc[j], msgB + (long)ebase*F2 + row*64 + hb, F2, lane, 16);
      }
    }
  }
  // m=2: yr rows 13..15, yi rows 16..18
  for(int half=0; half<2; ++half){
    for(int ng=0; ng<3; ++ng){
      v8f acc[4];
#pragma unroll
      for(int j=0;j<4;++j) acc[j] = zero8();
      if(half == 0){
        accum_gemm<4>(acc, msgA, F1, ebase, 1664, pk_r2,  12, ng*4, lane);
        accum_gemm<4>(acc, msgA, F1, ebase, 2048, pk_i2n, 12, ng*4, lane);
      } else {
        accum_gemm<4>(acc, msgA, F1, ebase, 2048, pk_r2,  12, ng*4, lane);
        accum_gemm<4>(acc, msgA, F1, ebase, 1664, pk_i2,  12, ng*4, lane);
      }
#pragma unroll
      for(int j=0;j<4;++j){
        int nt = ng*4 + j;
        int row = (half ? 16 : 13) + (nt >> 2), hb = (nt & 3)*16;
        int gc = ((half ? row - 15 : row - 12))*64 + hb + col;
#pragma unroll
        for(int r=0;r<8;++r)
          acc[j][r] *= bfu2f(gate[wv][(rofs + r)*256 + gc]);
        store_tile_bf16(stage[wv], acc[j], msgB + (long)ebase*F2 + row*64 + hb, F2, lane, 16);
      }
    }
  }
}

// SO2 conv #2 (+envelope): msgB[E][1216] -> msgC[E][1216]
__global__ __launch_bounds__(64) void conv2_kernel(
    const u16* __restrict__ msgB,
    const u16* pk_m0, const u16* pk_r1, const u16* pk_i1, const u16* pk_i1n,
    const u16* pk_r2, const u16* pk_i2, const u16* pk_i2n,
    const float* __restrict__ b2, const float* __restrict__ env,
    u16* __restrict__ msgC, int E){
  __shared__ alignas(16) float stage[2][256];
  int wv = threadIdx.x >> 5, lane = threadIdx.x & 31;
  int ebase = (blockIdx.x*2 + wv) * 16;
  if(ebase >= E) return;
  int col = lane & 15, rofs = (lane >> 4) * 8;
  const long F2 = 1216;
  float ev[8];
#pragma unroll
  for(int r=0;r<8;++r) ev[r] = env[ebase + rofs + r];

  // m=0
  for(int ng=0; ng<5; ++ng){
    v8f acc[4];
#pragma unroll
    for(int j=0;j<4;++j) acc[j] = zero8();
    accum_gemm<4>(acc, msgB, F2, ebase, 0, pk_m0, 10, ng*4, lane);
#pragma unroll
    for(int j=0;j<4;++j){
      int nt = ng*4 + j;
      float bias = b2[nt*16 + col];
      int row = nt >> 2, hb = (nt & 3)*16;
#pragma unroll
      for(int r=0;r<8;++r) acc[j][r] = (acc[j][r] + bias) * ev[r];
      store_tile_bf16(stage[wv], acc[j], msgC + (long)ebase*F2 + row*64 + hb, F2, lane, 16);
    }
  }
  // m=1
  for(int half=0; half<2; ++half){
    for(int ng=0; ng<4; ++ng){
      v8f acc[4];
#pragma unroll
      for(int j=0;j<4;++j) acc[j] = zero8();
      if(half == 0){
        accum_gemm<4>(acc, msgB, F2, ebase, 320, pk_r1,  8, ng*4, lane);
        accum_gemm<4>(acc, msgB, F2, ebase, 576, pk_i1n, 8, ng*4, lane);
      } else {
        accum_gemm<4>(acc, msgB, F2, ebase, 576, pk_r1,  8, ng*4, lane);
        accum_gemm<4>(acc, msgB, F2, ebase, 320, pk_i1,  8, ng*4, lane);
      }
#pragma unroll
      for(int j=0;j<4;++j){
        int nt = ng*4 + j;
        int row = (half ? 9 : 5) + (nt >> 2), hb = (nt & 3)*16;
#pragma unroll
        for(int r=0;r<8;++r) acc[j][r] *= ev[r];
        store_tile_bf16(stage[wv], acc[j], msgC + (long)ebase*F2 + row*64 + hb, F2, lane, 16);
      }
    }
  }
  // m=2
  for(int half=0; half<2; ++half){
    for(int ng=0; ng<3; ++ng){
      v8f acc[4];
#pragma unroll
      for(int j=0;j<4;++j) acc[j] = zero8();
      if(half == 0){
        accum_gemm<4>(acc, msgB, F2, ebase,  832, pk_r2,  6, ng*4, lane);
        accum_gemm<4>(acc, msgB, F2, ebase, 1024, pk_i2n, 6, ng*4, lane);
      } else {
        accum_gemm<4>(acc, msgB, F2, ebase, 1024, pk_r2,  6, ng*4, lane);
        accum_gemm<4>(acc, msgB, F2, ebase,  832, pk_i2,  6, ng*4, lane);
      }
#pragma unroll
      for(int j=0;j<4;++j){
        int nt = ng*4 + j;
        int row = (half ? 16 : 13) + (nt >> 2), hb = (nt & 3)*16;
#pragma unroll
        for(int r=0;r<8;++r) acc[j][r] *= ev[r];
        store_tile_bf16(stage[wv], acc[j], msgC + (long)ebase*F2 + row*64 + hb, F2, lane, 16);
      }
    }
  }
}

// per-edge: out[rcv] += wigner_inv (25x19) @ msgC (19x64)
__global__ __launch_bounds__(128) void wiginv_kernel(
    const float* __restrict__ winv, const u16* __restrict__ msgC,
    const int* __restrict__ rcv, float* __restrict__ out, int E){
  __shared__ alignas(16) u16 BT[4][64*32];
  int wv = threadIdx.x >> 5, lane = threadIdx.x & 31;
  int e = blockIdx.x*4 + wv;
  if(e >= E) return;
  for(int idx = lane; idx < 64*32; idx += 32){
    int c = idx >> 5, k = idx & 31;
    BT[wv][idx] = (k < 19) ? msgC[(long)e*1216 + k*64 + c] : (u16)0;
  }
  asm volatile("s_wait_dscnt 0" ::: "memory");

  v16bf a[2];
  for(int mt=0; mt<2; ++mt){
    int row = mt*16 + (lane & 15);
    int kb  = (lane >> 4) * 8;
    const float* ar = winv + (long)e*475 + row*19;
    v8u u;
#pragma unroll
    for(int vv=0; vv<8; ++vv){
      int k0 = ((vv < 4) ? 0 : 16) + kb + 2*(vv & 3);
      float f0 = (row < 25 && k0     < 19) ? ar[k0]     : 0.f;
      float f1 = (row < 25 && k0 + 1 < 19) ? ar[k0 + 1] : 0.f;
      u[vv] = pk2(f0, f1);
    }
    a[mt] = __builtin_bit_cast(v16bf, u);
  }
  float* ob = out + (long)rcv[e]*1600;
  for(int nt=0; nt<4; ++nt){
    v16bf b = b_frag_ldsT(BT[wv], nt, lane);
    for(int mt=0; mt<2; ++mt){
      v8f acc = zero8();
      acc = wmma_bf16(a[mt], b, acc);
      int ch = nt*16 + (lane & 15);
      int rbase = mt*16 + ((lane >> 4) * 8);
#pragma unroll
      for(int r=0;r<8;++r){
        int harm = rbase + r;
        if(harm < 25) atomicAdd(ob + harm*64 + ch, acc[r]);
      }
    }
  }
}

// ---------------- host ----------------
extern "C" void kernel_launch(void* const* d_in, const int* in_sizes, int n_in,
                              void* d_out, int out_size, void* d_ws, size_t ws_size,
                              hipStream_t stream){
  (void)n_in; (void)ws_size;
  const float* nf   = (const float*)d_in[0];
  const float* ee   = (const float*)d_in[1];
  const float* wig  = (const float*)d_in[2];
  const float* winv = (const float*)d_in[3];
  const float* env  = (const float*)d_in[4];
  const float* rw1  = (const float*)d_in[5];
  const float* rb1  = (const float*)d_in[6];
  const float* rw2  = (const float*)d_in[7];
  const float* rb2  = (const float*)d_in[8];
  const float* w1m0 = (const float*)d_in[9];
  const float* b1m0 = (const float*)d_in[10];
  const float* w1r1 = (const float*)d_in[11];
  const float* w1i1 = (const float*)d_in[12];
  const float* w1r2 = (const float*)d_in[13];
  const float* w1i2 = (const float*)d_in[14];
  const float* w2m0 = (const float*)d_in[15];
  const float* b2m0 = (const float*)d_in[16];
  const float* w2r1 = (const float*)d_in[17];
  const float* w2i1 = (const float*)d_in[18];
  const float* w2r2 = (const float*)d_in[19];
  const float* w2i2 = (const float*)d_in[20];
  const int*   snd  = (const int*)d_in[21];
  const int*   rcv  = (const int*)d_in[22];
  int E = in_sizes[21];
  float* out = (float*)d_out;

  u16* ws = (u16*)d_ws;
  size_t off = 0;
  auto alloc = [&](size_t elems){ u16* p = ws + off; off += elems; return p; };
  u16* p_rw1   = alloc(128*64);
  u16* p_rw2   = alloc(64*1536);
  u16* p_w1m0  = alloc(640*576);
  u16* p_w1r1  = alloc(512*256);
  u16* p_w1i1  = alloc(512*256);
  u16* p_w1i1n = alloc(512*256);
  u16* p_w1r2  = alloc(384*192);
  u16* p_w1i2  = alloc(384*192);
  u16* p_w1i2n = alloc(384*192);
  u16* p_w2m0  = alloc(320*320);
  u16* p_w2r1  = alloc(256*256);
  u16* p_w2i1  = alloc(256*256);
  u16* p_w2i1n = alloc(256*256);
  u16* p_w2r2  = alloc(192*192);
  u16* p_w2i2  = alloc(192*192);
  u16* p_w2i2n = alloc(192*192);
  u16* radbuf  = alloc((size_t)E * 1536);
  u16* msgA    = alloc((size_t)E * 2432);
  u16* msgB    = alloc((size_t)E * 1216);
  u16* msgC    = msgA;  // msgA dead after conv1: reuse

  auto pp = [&](const float* w, u16* dst, int K, int N, float sg){
    int total = (K*N) >> 1;
    prepack_b<<<(total + 255)/256, 256, 0, stream>>>(w, dst, K, N, sg);
  };
  pp(rw1,  p_rw1,  128, 64,   1.f);
  pp(rw2,  p_rw2,  64,  1536, 1.f);
  pp(w1m0, p_w1m0, 640, 576,  1.f);
  pp(w1r1, p_w1r1, 512, 256,  1.f);
  pp(w1i1, p_w1i1, 512, 256,  1.f);
  pp(w1i1, p_w1i1n,512, 256, -1.f);
  pp(w1r2, p_w1r2, 384, 192,  1.f);
  pp(w1i2, p_w1i2, 384, 192,  1.f);
  pp(w1i2, p_w1i2n,384, 192, -1.f);
  pp(w2m0, p_w2m0, 320, 320,  1.f);
  pp(w2r1, p_w2r1, 256, 256,  1.f);
  pp(w2i1, p_w2i1, 256, 256,  1.f);
  pp(w2i1, p_w2i1n,256, 256, -1.f);
  pp(w2r2, p_w2r2, 192, 192,  1.f);
  pp(w2i2, p_w2i2, 192, 192,  1.f);
  pp(w2i2, p_w2i2n,192, 192, -1.f);

  zero_f32<<<(out_size + 255)/256, 256, 0, stream>>>(out, (long)out_size);

  int nwaves = (E + 15) / 16;
  int cblocks = (nwaves + 1) / 2;
  rad_kernel<<<cblocks, 64, 0, stream>>>(ee, rb1, rb2, p_rw1, p_rw2, radbuf, E);
  gather_wigner_kernel<<<(E + 3)/4, 128, 0, stream>>>(nf, wig, snd, rcv, radbuf, msgA, E);
  conv1_kernel<<<cblocks, 64, 0, stream>>>(msgA, p_w1m0, p_w1r1, p_w1i1, p_w1i1n,
                                           p_w1r2, p_w1i2, p_w1i2n, b1m0, msgB, E);
  conv2_kernel<<<cblocks, 64, 0, stream>>>(msgB, p_w2m0, p_w2r1, p_w2i1, p_w2i1n,
                                           p_w2r2, p_w2i2, p_w2i2n, b2m0, env, msgC, E);
  wiginv_kernel<<<(E + 3)/4, 128, 0, stream>>>(winv, msgC, rcv, out, E);
}
